// MultiheadAttentionOperator_80547816669371
// MI455X (gfx1250) — matrix operand order repaired and checked by reading, not compiled, not run on hardware
//
#include <hip/hip_runtime.h>
#include <math.h>

// ---------------------------------------------------------------------------
// Spectral multihead attention for MI455X (gfx1250, wave32, WMMA 16x16x32 f16)
// All DFT / mode-mixing / inverse-transform math is f16 WMMA with f32 accum.
// All WMMA operands are staged in per-lane-contiguous layouts so LDS traffic
// is ds_load_b128 / ds_store_b128 (no scalar u16 gathers).
// ---------------------------------------------------------------------------

typedef __attribute__((ext_vector_type(16))) _Float16 v16h;
typedef __attribute__((ext_vector_type(8)))  _Float16 v8h;
typedef __attribute__((ext_vector_type(8)))  float    v8f;

#define PI_OVER_32 0.09817477042468103f

__device__ __forceinline__ v8f wmma32(v16h a, v16h b, v8f c) {
  return __builtin_amdgcn_wmma_f32_16x16x32_f16(
      false, a, false, b, (short)0, c, false, false);
}

// A 16x32 (MxK), row-major [M][K]: lane m holds runs [koff..koff+7] and
// [16+koff..16+koff+7]  (koff = 0 / 8 for lane<16 / >=16)  -> two b128 loads.
__device__ __forceinline__ v16h loadA16x32(const _Float16* p, int lda) {
  int lane = threadIdx.x & 31;
  int row  = lane & 15;
  int koff = (lane & 16) ? 8 : 0;
  const _Float16* s = p + row * lda + koff;
  v8h a0 = *(const v8h*)(s);
  v8h a1 = *(const v8h*)(s + 16);
  return __builtin_shufflevector(a0, a1, 0,1,2,3,4,5,6,7,8,9,10,11,12,13,14,15);
}

// B 32x16 (KxN) stored N-major [N][K]: lane n holds K runs [koff..koff+7] and
// [16+koff..16+koff+7] (koff = 0 / 8)  -> two b128 loads. ldn must be a multiple of 8.
__device__ __forceinline__ v16h loadB32x16T(const _Float16* p, int ldn) {
  int lane = threadIdx.x & 31;
  int col  = lane & 15;
  int koff = (lane & 16) ? 8 : 0;
  const _Float16* s = p + col * ldn + koff;
  v8h b0 = *(const v8h*)(s);
  v8h b1 = *(const v8h*)(s + 16);
  return __builtin_shufflevector(b0, b1, 0,1,2,3,4,5,6,7,8,9,10,11,12,13,14,15);
}

// C/D 16x16: VGPR r -> M = r + (lane>=16 ? 8 : 0), N = lane&15.
__device__ __forceinline__ int cRow(int r) {
  return r + ((threadIdx.x & 16) ? 8 : 0);
}

__device__ __forceinline__ void copy16B(uint4* dst, const uint4* src, int n,
                                        int tid, int nthr) {
  for (int i = tid; i < n; i += nthr) dst[i] = src[i];
}

// ---------------------------------------------------------------------------
// K_init: twiddles + Gaussian spectrum
// ---------------------------------------------------------------------------
__global__ void k_init_tw(_Float16* Ftw,   // [64][64] rows 0-31 Re, 32-63 Im of e^{-i2pi ky h/64}
                          _Float16* Fwre, _Float16* Fwim, _Float16* FwimN, // [16][64]
                          _Float16* Ere, _Float16* Eim, _Float16* EimN,    // [64][32] e^{+i2pi ky h/64}
                          _Float16* TBt,   // transposed [w=64][k=32]: k<16: c*cos/4096, k>=16: -c*sin/4096
                          float2* kfc)     // [512] conj(rfft2(rolled gauss)) at modes
{
  int tid = threadIdx.x;
  for (int i = tid; i < 4096; i += 256) {
    int r = i >> 6, h = i & 63, idx = r & 31;
    int ky = (idx < 16) ? idx : idx + 32;
    float a = (float)((ky * h) & 63) * PI_OVER_32;
    Ftw[i] = (_Float16)((r < 32) ? cosf(a) : -sinf(a));
  }
  for (int i = tid; i < 1024; i += 256) {
    int kx = i >> 6, w = i & 63;
    float a = (float)((kx * w) & 63) * PI_OVER_32;
    Fwre[i] = (_Float16)cosf(a);
    Fwim[i] = (_Float16)(-sinf(a));
    FwimN[i] = (_Float16)sinf(a);
  }
  for (int i = tid; i < 2048; i += 256) {
    int h = i >> 5, idx = i & 31;
    int ky = (idx < 16) ? idx : idx + 32;
    float a = (float)((ky * h) & 63) * PI_OVER_32;
    Ere[i] = (_Float16)cosf(a);
    Eim[i] = (_Float16)sinf(a);
    EimN[i] = (_Float16)(-sinf(a));
  }
  for (int i = tid; i < 2048; i += 256) {
    int w = i >> 5, r = i & 31, kx = r & 15;
    float a = (float)((kx * w) & 63) * PI_OVER_32;
    float c = ((kx == 0) ? 1.0f : 2.0f) * (1.0f / 4096.0f);
    TBt[i] = (_Float16)((r < 16) ? c * cosf(a) : -c * sinf(a));
  }
  for (int m = tid; m < 512; m += 256) {
    int kyI = m >> 4;
    int ky = (kyI < 16) ? kyI : kyI + 32;
    int kx = m & 15;
    float S = 0.0f, yr = 0.0f, yi = 0.0f, xr = 0.0f, xi = 0.0f;
    for (int h = 0; h < 64; ++h) {
      float dh = (float)h - 31.5f;
      float g = expf(-0.5f * dh * dh);
      S += g;
      int js = (h + 32) & 63;  // roll by -32
      float ay = (float)((ky * js) & 63) * PI_OVER_32;
      float ax = (float)((kx * js) & 63) * PI_OVER_32;
      yr += g * cosf(ay); yi -= g * sinf(ay);
      xr += g * cosf(ax); xi -= g * sinf(ax);
    }
    float pr = yr * xr - yi * xi;
    float pi = yr * xi + yi * xr;
    float invZ = 1.0f / (S * S);
    kfc[m] = make_float2(pr * invZ, -pi * invZ);  // conj
  }
}

// ---------------------------------------------------------------------------
// K_pack: weights -> per-mode f16 planes, TRANSPOSED: [m][oh][i], oh = o*4+h
// ---------------------------------------------------------------------------
struct WPtrs { const float2* p[6]; };

__global__ void k_pack(WPtrs wp, _Float16* Wre, _Float16* Wim) {
  int t6 = blockIdx.x >> 8;       // 0..5 : q1,q2,k1,k2,v1,v2
  int mlocal = blockIdx.x & 255;
  int pt = t6 >> 1;               // 0=q,1=k,2=v
  int isBot = t6 & 1;
  int m = mlocal + (isBot ? 256 : 0);
  int x = (m >> 4) & 15;
  int y = m & 15;
  const float2* W = wp.p[t6];
  size_t dbase = ((size_t)(pt * 512 + m)) * 4096;
  for (int e = threadIdx.x; e < 4096; e += 256) {
    int oh = e >> 6, i = e & 63;
    int o = oh >> 2, hh = oh & 3;
    size_t s = ((size_t)((i * 16 + o) * 16 + x) * 16 + y) * 4 + hh;
    float2 v = W[s];
    Wre[dbase + e] = (_Float16)v.x;
    Wim[dbase + e] = (_Float16)v.y;
  }
}

// ---------------------------------------------------------------------------
// K_copy: output[:, :, :64] = x
// ---------------------------------------------------------------------------
__global__ void k_copy_x(const float4* z4, float4* out4) {
  size_t idx = (size_t)blockIdx.x * 256 + threadIdx.x;  // 64 * 65536 total
  size_t bt = idx >> 16, j = idx & 65535;
  out4[bt * 131072 + j] = z4[bt * 131072 + j];
}

// ---------------------------------------------------------------------------
// K1: row DFT: T1[ky,*] = sum_h F[ky,h]*img[h,*]; GEMM 64 x 256cols x K=64.
//     B staged transposed [col][h] (ld 72); T1 written TRANSPOSED [c][w].
// ---------------------------------------------------------------------------
__global__ void __launch_bounds__(256) k1_rowdft(const float* z, const _Float16* Ftw,
                                                 _Float16* T1re, _Float16* T1im) {
  int bts = blockIdx.x;                  // (b*16+t)*2 + src
  int colBase = blockIdx.y << 8;
  int bt = bts >> 1, src = bts & 1;
  __shared__ __align__(16) _Float16 sA[4096];       // 64x64 twiddles, [M][K]
  __shared__ __align__(16) _Float16 sBT[256 * 72];  // [col][h], padded
  int tid = threadIdx.x;
  copy16B((uint4*)sA, (const uint4*)Ftw, 512, tid, 256);
  const float* rowbase = z + ((size_t)(bt * 128 + src * 64)) * 4096 + colBase;
  for (int i = tid; i < 4096; i += 256) {
    int h = i >> 6, q4 = i & 63;
    float4 v = ((const float4*)(rowbase + (size_t)h * 4096))[q4];
    int c0 = q4 * 4;
    sBT[(c0 + 0) * 72 + h] = (_Float16)v.x;
    sBT[(c0 + 1) * 72 + h] = (_Float16)v.y;
    sBT[(c0 + 2) * 72 + h] = (_Float16)v.z;
    sBT[(c0 + 3) * 72 + h] = (_Float16)v.w;
  }
  __syncthreads();
  int wave = tid >> 5, lane = tid & 31;
#pragma unroll
  for (int i = 0; i < 8; ++i) {
    int tt = wave * 8 + i;
    int rt = tt >> 4, ct = tt & 15;
    v8f acc = {};
#pragma unroll
    for (int ks = 0; ks < 2; ++ks) {
      v16h a = loadA16x32(sA + rt * 16 * 64 + ks * 32, 64);
      v16h b = loadB32x16T(sBT + ct * 16 * 72 + ks * 32, 72);
      acc = wmma32(a, b, acc);
    }
    int colT = colBase + ct * 16 + (lane & 15);
    int cc = colT & 63, ww = colT >> 6;
#pragma unroll
    for (int r = 0; r < 8; ++r) {
      int gr = rt * 16 + cRow(r);
      size_t off = ((size_t)(bts * 32 + (gr & 31))) * 4096 + cc * 64 + ww;
      _Float16 hv = (_Float16)acc[r];
      if (gr < 32) T1re[off] = hv; else T1im[off] = hv;
    }
  }
}

// ---------------------------------------------------------------------------
// K2: col DFT (complex): modes[kx,c] = sum_w Fw[kx,w]*T1[ky][w,c]
//     B = T1t[ky] already [n=c][k=w]; writes A planes Are/Aim[src][m][bt][c]
// ---------------------------------------------------------------------------
__global__ void __launch_bounds__(128) k2_coldft(const _Float16* T1re, const _Float16* T1im,
                                                 const _Float16* Fwre, const _Float16* Fwim,
                                                 const _Float16* FwimN,
                                                 _Float16* Are, _Float16* Aim) {
  int bts = blockIdx.x, kyI = blockIdx.y;
  __shared__ __align__(16) _Float16 sBre[4096], sBim[4096];     // [c][w]
  __shared__ __align__(16) _Float16 sAre[1024], sAim[1024], sAimN[1024];
  int tid = threadIdx.x;
  size_t tb = ((size_t)(bts * 32 + kyI)) * 4096;
  __builtin_prefetch(T1re + tb, 0, 1);
  __builtin_prefetch(T1im + tb, 0, 1);
  copy16B((uint4*)sBre, (const uint4*)(T1re + tb), 512, tid, 128);
  copy16B((uint4*)sBim, (const uint4*)(T1im + tb), 512, tid, 128);
  copy16B((uint4*)sAre, (const uint4*)Fwre, 128, tid, 128);
  copy16B((uint4*)sAim, (const uint4*)Fwim, 128, tid, 128);
  copy16B((uint4*)sAimN, (const uint4*)FwimN, 128, tid, 128);
  __syncthreads();
  int ct = tid >> 5, lane = tid & 31;
  v8f cre = {}, cim = {};
#pragma unroll
  for (int ks = 0; ks < 2; ++ks) {
    v16h aRe = loadA16x32(sAre + ks * 32, 64);
    v16h aIm = loadA16x32(sAim + ks * 32, 64);
    v16h aImN = loadA16x32(sAimN + ks * 32, 64);
    v16h bRe = loadB32x16T(sBre + ct * 16 * 64 + ks * 32, 64);
    v16h bIm = loadB32x16T(sBim + ct * 16 * 64 + ks * 32, 64);
    cre = wmma32(aRe, bRe, cre); cre = wmma32(aImN, bIm, cre);
    cim = wmma32(aRe, bIm, cim); cim = wmma32(aIm, bRe, cim);
  }
  int c = ct * 16 + (lane & 15);
  int src = bts & 1, bt = bts >> 1;
#pragma unroll
  for (int r = 0; r < 8; ++r) {
    int kx = cRow(r);
    int m = kyI * 16 + kx;
    size_t off = ((size_t)(src * 512 + m) * 64 + bt) * 64 + c;
    Are[off] = (_Float16)cre[r];
    Aim[off] = (_Float16)cim[r];
  }
}

// ---------------------------------------------------------------------------
// K3: per-mode mixing: OUT[bt,oh] = sum_i A_m[bt,i]*W_m[i,oh] (complex 64^3)
//     W planes are [m][oh][i] (N-major). t3: 0=Q, 1=K, 2=V(*conj Gaussian)
// ---------------------------------------------------------------------------
__global__ void __launch_bounds__(256) k3_mix(const _Float16* Are, const _Float16* Aim,
                                              const _Float16* Wre, const _Float16* Wim,
                                              const float2* kfc,
                                              float2* Qc, float2* Kc, float2* Vc) {
  int m = blockIdx.x, t3 = blockIdx.y;
  __shared__ __align__(16) _Float16 sAre[4096], sAim[4096];           // [bt][i]
  __shared__ __align__(16) _Float16 sWre[4096], sWim[4096], sWimN[4096]; // [oh][i]
  int tid = threadIdx.x;
  int src = (t3 == 2) ? 1 : 0;
  size_t ab = ((size_t)(src * 512 + m)) * 4096;
  size_t wb = ((size_t)(t3 * 512 + m)) * 4096;
  __builtin_prefetch(Are + ab, 0, 1);
  __builtin_prefetch(Wre + wb, 0, 1);
  __builtin_prefetch(Wim + wb, 0, 1);
  copy16B((uint4*)sAre, (const uint4*)(Are + ab), 512, tid, 256);
  copy16B((uint4*)sAim, (const uint4*)(Aim + ab), 512, tid, 256);
  copy16B((uint4*)sWre, (const uint4*)(Wre + wb), 512, tid, 256);
  for (int i = tid; i < 512; i += 256) {   // Im plane + sign-flipped copy
    uint4 v = ((const uint4*)(Wim + wb))[i];
    ((uint4*)sWim)[i] = v;
    uint4 nv;
    nv.x = v.x ^ 0x80008000u; nv.y = v.y ^ 0x80008000u;
    nv.z = v.z ^ 0x80008000u; nv.w = v.w ^ 0x80008000u;
    ((uint4*)sWimN)[i] = nv;
  }
  __syncthreads();
  int wave = tid >> 5, lane = tid & 31;
  float2* dstT = (t3 == 0) ? Qc : ((t3 == 1) ? Kc : Vc);
  float2 kv = kfc[m];
#pragma unroll
  for (int i = 0; i < 2; ++i) {
    int tt = wave * 2 + i;
    int rt = tt >> 2, ct = tt & 3;
    v8f cre = {}, cim = {};
#pragma unroll
    for (int ks = 0; ks < 2; ++ks) {
      v16h aRe = loadA16x32(sAre + rt * 16 * 64 + ks * 32, 64);
      v16h aIm = loadA16x32(sAim + rt * 16 * 64 + ks * 32, 64);
      v16h bRe = loadB32x16T(sWre + ct * 16 * 64 + ks * 32, 64);
      v16h bIm = loadB32x16T(sWim + ct * 16 * 64 + ks * 32, 64);
      v16h bImN = loadB32x16T(sWimN + ct * 16 * 64 + ks * 32, 64);
      cre = wmma32(aRe, bRe, cre); cre = wmma32(aIm, bImN, cre);
      cim = wmma32(aRe, bIm, cim); cim = wmma32(aIm, bRe, cim);
    }
#pragma unroll
    for (int r = 0; r < 8; ++r) {
      int bt = rt * 16 + cRow(r);
      int oh = ct * 16 + (lane & 15);
      float xre = cre[r], xim = cim[r];
      if (t3 == 2) {  // multiply by conj(Gaussian spectrum)
        float nr = xre * kv.x - xim * kv.y;
        float ni = xre * kv.y + xim * kv.x;
        xre = nr; xim = ni;
      }
      dstT[((size_t)bt * 64 + oh) * 512 + m] = make_float2(xre, xim);
    }
  }
}

// ---------------------------------------------------------------------------
// K4: gram in mode space (Parseval + kx=0 Hermitian pair corrections)
// ---------------------------------------------------------------------------
__global__ void __launch_bounds__(256) k4_gram(const float2* Qc, const float2* Kc, float* kxg) {
  int bid = blockIdx.x;
  int qq = bid & 15, p = (bid >> 4) & 15, n = (bid >> 8) & 3, b = bid >> 10;
  int tid = threadIdx.x;
  float part = 0.0f;
  for (int idx = tid; idx < 8192; idx += 256) {
    int d = idx >> 9, m = idx & 511;
    int oh = d * 4 + n;
    float2 qv = Qc[((size_t)((b * 16 + p) * 64 + oh)) * 512 + m];
    float2 kv = Kc[((size_t)((b * 16 + qq) * 64 + oh)) * 512 + m];
    float w = ((m & 15) == 0) ? 0.5f : 2.0f;   // c^2/2
    part += w * (qv.x * kv.x + qv.y * kv.y);
  }
  for (int ci = tid; ci < 496; ci += 256) {  // kx=0 pair terms (ky=48 pairs w/ zero mode)
    int d = ci / 31, rr = ci % 31;
    int r = (rr < 16) ? rr : rr + 1;
    int rf = (r == 0) ? 0 : (32 - r);
    int oh = d * 4 + n;
    float2 qv = Qc[((size_t)((b * 16 + p) * 64 + oh)) * 512 + r * 16];
    float2 kv = Kc[((size_t)((b * 16 + qq) * 64 + oh)) * 512 + rf * 16];
    part += 0.5f * (qv.x * kv.x - qv.y * kv.y);
  }
  __shared__ float sred[256];
  sred[tid] = part;
  __syncthreads();
  for (int s = 128; s > 0; s >>= 1) {
    if (tid < s) sred[tid] += sred[tid + s];
    __syncthreads();
  }
  if (tid == 0) kxg[bid] = sred[0] * (1.0f / 16777216.0f);
}

// ---------------------------------------------------------------------------
// K5: ofm[b,n,p,d,m] = sum_q kx[b,n,p,q] * V[b,q,d*4+n,m]
// ---------------------------------------------------------------------------
__global__ void __launch_bounds__(256) k5_kxmix(const float* kxg, const float2* Vc, float2* ofm) {
  int bid = blockIdx.x;
  int d = bid & 15, n = (bid >> 8) & 3, b = bid >> 10;
  int tid = threadIdx.x;
  __shared__ float sk[16];
  if (tid < 16) sk[tid] = kxg[(bid & ~15) + tid];
  __syncthreads();
  int oh = d * 4 + n;
  for (int m = tid; m < 512; m += 256) {
    float ar = 0.0f, ai = 0.0f;
#pragma unroll
    for (int q = 0; q < 16; ++q) {
      float2 v = Vc[((size_t)((b * 16 + q) * 64 + oh)) * 512 + m];
      ar += sk[q] * v.x;
      ai += sk[q] * v.y;
    }
    ofm[(size_t)bid * 512 + m] = make_float2(ar, ai);
  }
}

// ---------------------------------------------------------------------------
// K6: inverse transform + reshape scatter:
//     out[b, n*4+p/4, 64+(p%4)*16+h/4, (h%4)*16+w/4, (w%4)*16+d]
// ---------------------------------------------------------------------------
__global__ void __launch_bounds__(128) k6_inv(const float2* ofm,
                                              const _Float16* Ere, const _Float16* Eim,
                                              const _Float16* EimN, const _Float16* TBt,
                                              float* out) {
  int bid = blockIdx.x;
  int d = bid & 15, p = (bid >> 4) & 15, n = (bid >> 8) & 3, b = bid >> 10;
  int tid = threadIdx.x;
  __shared__ __align__(16) _Float16 sMreT[512], sMimT[512];  // [kx][ky], ld 32
  __shared__ __align__(16) _Float16 sA2[64 * 32];            // [h][Gre|Gim]
  for (int i = tid; i < 512; i += 128) {
    float2 v = ofm[(size_t)bid * 512 + i];
    int ky = i >> 4, kx = i & 15;
    sMreT[kx * 32 + ky] = (_Float16)v.x;
    sMimT[kx * 32 + ky] = (_Float16)v.y;
  }
  __syncthreads();
  int rt = tid >> 5, lane = tid & 31;
  // Stage A: G[h,kx] = sum_ky E[h,ky]*M[ky,kx]  (M=64,N=16,K=32, complex)
  {
    v16h aRe = loadA16x32(Ere + rt * 16 * 32, 32);
    v16h aIm = loadA16x32(Eim + rt * 16 * 32, 32);
    v16h aImN = loadA16x32(EimN + rt * 16 * 32, 32);
    v16h bRe = loadB32x16T(sMreT, 32);
    v16h bIm = loadB32x16T(sMimT, 32);
    v8f gre = {}, gim = {};
    gre = wmma32(aRe, bRe, gre); gre = wmma32(aImN, bIm, gre);
    gim = wmma32(aRe, bIm, gim); gim = wmma32(aIm, bRe, gim);
#pragma unroll
    for (int r = 0; r < 8; ++r) {
      int h = rt * 16 + cRow(r);
      int c = lane & 15;
      sA2[h * 32 + c] = (_Float16)gre[r];
      sA2[h * 32 + 16 + c] = (_Float16)gim[r];
    }
  }
  __syncthreads();
  // Stage B: img = [Gre|Gim](64x32) @ [c*cos; -c*sin]/4096, B stored [w][k]
  v16h a2 = loadA16x32(sA2 + rt * 16 * 32, 32);
#pragma unroll
  for (int ct = 0; ct < 4; ++ct) {
    v16h bT = loadB32x16T(TBt + ct * 16 * 32, 32);
    v8f img = {};
    img = wmma32(a2, bT, img);
#pragma unroll
    for (int r = 0; r < 8; ++r) {
      int h = rt * 16 + cRow(r);
      int w = ct * 16 + (lane & 15);
      int t = n * 4 + (p >> 2);
      int hp = (p & 3) * 16 + (h >> 2);
      int wp = (h & 3) * 16 + (w >> 2);
      int cp = (w & 3) * 16 + d;
      size_t off = (((size_t)(b * 16 + t) * 128 + 64 + hp) * 64 + wp) * 64 + cp;
      out[off] = img[r];
    }
  }
}

// ---------------------------------------------------------------------------
// Host launcher
// ---------------------------------------------------------------------------
extern "C" void kernel_launch(void* const* d_in, const int* in_sizes, int n_in,
                              void* d_out, int out_size, void* d_ws, size_t ws_size,
                              hipStream_t stream) {
  (void)in_sizes; (void)n_in; (void)out_size; (void)ws_size;
  const float* z = (const float*)d_in[0];

  char* base = (char*)d_ws;
  size_t off = 0;
  auto take = [&](size_t bytes) -> void* {
    off = (off + 255) & ~(size_t)255;
    void* p = base + off;
    off += bytes;
    return p;
  };

  _Float16* Ftw   = (_Float16*)take(4096 * 2);
  _Float16* Fwre  = (_Float16*)take(1024 * 2);
  _Float16* Fwim  = (_Float16*)take(1024 * 2);
  _Float16* FwimN = (_Float16*)take(1024 * 2);
  _Float16* Ere   = (_Float16*)take(2048 * 2);
  _Float16* Eim   = (_Float16*)take(2048 * 2);
  _Float16* EimN  = (_Float16*)take(2048 * 2);
  _Float16* TBt   = (_Float16*)take(2048 * 2);
  float2*   kfc   = (float2*)take(512 * 8);
  _Float16* Wre   = (_Float16*)take((size_t)3 * 512 * 4096 * 2);
  _Float16* Wim   = (_Float16*)take((size_t)3 * 512 * 4096 * 2);
  _Float16* Are   = (_Float16*)take((size_t)2 * 512 * 4096 * 2);
  _Float16* Aim   = (_Float16*)take((size_t)2 * 512 * 4096 * 2);
  _Float16* T1re  = (_Float16*)take((size_t)128 * 32 * 4096 * 2);
  _Float16* T1im  = (_Float16*)take((size_t)128 * 32 * 4096 * 2);
  float2*   Qc    = (float2*)take((size_t)64 * 64 * 512 * 8);
  float2*   Kc    = (float2*)take((size_t)64 * 64 * 512 * 8);
  float2*   Vc    = (float2*)take((size_t)64 * 64 * 512 * 8);
  float*    kxg   = (float*)take(4096 * 4);
  float2*   ofm   = (float2*)take((size_t)4096 * 512 * 8);

  WPtrs wp;
  for (int i = 0; i < 6; ++i) wp.p[i] = (const float2*)d_in[1 + i];

  k_init_tw<<<1, 256, 0, stream>>>(Ftw, Fwre, Fwim, FwimN, Ere, Eim, EimN, TBt, kfc);
  k_pack<<<dim3(1536), 256, 0, stream>>>(wp, Wre, Wim);
  k_copy_x<<<16384, 256, 0, stream>>>((const float4*)z, (float4*)d_out);
  k1_rowdft<<<dim3(128, 16), 256, 0, stream>>>(z, Ftw, T1re, T1im);
  k2_coldft<<<dim3(128, 32), 128, 0, stream>>>(T1re, T1im, Fwre, Fwim, FwimN, Are, Aim);
  k3_mix<<<dim3(512, 3), 256, 0, stream>>>(Are, Aim, Wre, Wim, kfc, Qc, Kc, Vc);
  k4_gram<<<4096, 256, 0, stream>>>(Qc, Kc, kxg);
  k5_kxmix<<<4096, 256, 0, stream>>>(kxg, Vc, ofm);
  k6_inv<<<4096, 128, 0, stream>>>(ofm, Ere, Eim, EimN, TBt, (float*)d_out);
}
